// SpikeGatedIncidence_43739946942800
// MI455X (gfx1250) — matrix operand order repaired and checked by reading, not compiled, not run on hardware
//
#include <hip/hip_runtime.h>
#include <hip/hip_bf16.h>

#define BB 8
#define NN 8192
#define EE 1024
#define DD 256
#define K2 512            // 2*D
#define WROWS 80          // 64 e-rows + 16 membrane-tile rows (row 64 = w_m)
#define ASTRIDE 520       // ushorts per LDS row (1040 B) -> conflict-free 16-row fragment reads

typedef __attribute__((ext_vector_type(16))) __bf16 v16bf;
typedef __attribute__((ext_vector_type(8)))  float  v8f;
typedef __attribute__((ext_vector_type(4)))  float  v4f;
typedef __attribute__((ext_vector_type(4)))  unsigned int v4u;
typedef __attribute__((ext_vector_type(2)))  unsigned int v2u;

__device__ __forceinline__ unsigned short f2bf(float f) {
  // round-to-nearest-even f32 -> bf16 (bit manipulation; no reliance on cvt builtins)
  unsigned u = __builtin_bit_cast(unsigned, f);
  unsigned r = u + 0x7FFFu + ((u >> 16) & 1u);
  return (unsigned short)(r >> 16);
}

__global__ void zero_kernel(float* __restrict__ p, int n) {
  int i = blockIdx.x * blockDim.x + threadIdx.x;
  if (i < n) p[i] = 0.0f;
}

// one thread per (row, 4-float chunk): atomic scatter-add of obs into sums[b, idx[row], :]
__global__ void scatter_kernel(const float* __restrict__ obs,
                               const int* __restrict__ idx,
                               float* __restrict__ sums,
                               float* __restrict__ counts) {
  int u = blockIdx.x * blockDim.x + threadIdx.x;     // B*N*64 threads
  int row = u >> 6;                                  // 0 .. B*N-1
  int c4  = u & 63;
  int b   = row >> 13;                               // / N
  int e   = idx[row];
  v4f o = *(const v4f*)(obs + (size_t)row * DD + c4 * 4);
  float* dst = sums + ((size_t)(b * EE + e)) * DD + c4 * 4;
  atomicAdd(dst + 0, o.x);
  atomicAdd(dst + 1, o.y);
  atomicAdd(dst + 2, o.z);
  atomicAdd(dst + 3, o.w);
  if (c4 == 0) atomicAdd(counts + b * EE + e, 1.0f);
}

// inline-asm LDS load: compiler cannot sink/rematerialize these
#define DSL(dst, addr, off) \
  asm volatile("ds_load_b128 %0, %1 offset:%c2" : "=v"(dst) : "v"(addr), "i"(off))

__global__ __launch_bounds__(128)
void fused_kernel(const float* __restrict__ obs,
                  const float* __restrict__ mask,
                  const int*   __restrict__ idx,
                  const float* __restrict__ w_m,
                  const float* __restrict__ b_m,
                  const float* __restrict__ w_e,
                  const float* __restrict__ b_e,
                  const float* __restrict__ sums,
                  const float* __restrict__ counts,
                  float* __restrict__ g_out,
                  float* __restrict__ e_out) {
  __shared__ unsigned short wlds[WROWS * ASTRIDE];        // 83,200 B  (bf16 weights, padded)
  __shared__ unsigned short atile[4][16 * ASTRIDE];       // 66,560 B  (bf16 A tiles, 1/wave)
  __shared__ float be_s[64];
  __shared__ float gvals[4][16];

  const int tid  = threadIdx.x;
  const int wave = tid >> 5;
  const int lane = tid & 31;

  // ---- preload weights into LDS as bf16: rows 0..63 = w_e, row 64 = w_m, 65..79 = 0 ----
  for (int i = tid; i < WROWS * K2; i += blockDim.x) {
    int rrow = i >> 9;            // / 512
    int col  = i & 511;
    float v = 0.0f;
    if (rrow < 64)       v = w_e[rrow * K2 + col];
    else if (rrow == 64) v = w_m[col];
    wlds[rrow * ASTRIDE + col] = f2bf(v);
  }
  if (tid < 64) be_s[tid] = b_e[tid];
  __syncthreads();

  const float bm = b_m[0];
  unsigned short* aw = &atile[wave][0];
  const int hi   = lane >> 4;      // 0 / 1 (half-wave)
  const int lrow = lane & 15;

  // per-lane 32-bit LDS byte addresses for the fragment streams (K-step 0)
  // (low 32 bits of a generic pointer to LDS == LDS byte offset per aperture rule)
  const unsigned aaddr = (unsigned)(unsigned long long)(aw + lrow * ASTRIDE + hi * 16);
  unsigned baddr[5];
#pragma unroll
  for (int nt = 0; nt < 5; ++nt)
    baddr[nt] = (unsigned)(unsigned long long)
                (wlds + (nt * 16 + lrow) * ASTRIDE + hi * 16);

  // 4096 tiles of 16 rows; 256 blocks * 4 waves -> exactly 4 tiles per wave
  for (int t = blockIdx.x * 4 + wave; t < (BB * NN) / 16; t += gridDim.x * 4) {
    // ---- build A tile (16 x 512 bf16): [obs | obs - ctx] ----
    for (int r = 0; r < 16; ++r) {
      const size_t gr = (size_t)t * 16 + r;
      const int b = (int)(gr >> 13);
      const int e = idx[gr];
      const float cnt = counts[b * EE + e];
      const float inv = 1.0f / fmaxf(cnt, 1.0f);
      const float* orow = obs + gr * DD;
      const float* srow = sums + (size_t)(b * EE + e) * DD;
#pragma unroll
      for (int p = 0; p < 2; ++p) {
        int c4 = lane + p * 32;                      // float4 index 0..63
        v4f o = *(const v4f*)(orow + c4 * 4);
        v4f s = *(const v4f*)(srow + c4 * 4);
        v4f d = o - s * inv;
        v2u po, pd;
        po.x = (unsigned)f2bf(o.x) | ((unsigned)f2bf(o.y) << 16);
        po.y = (unsigned)f2bf(o.z) | ((unsigned)f2bf(o.w) << 16);
        pd.x = (unsigned)f2bf(d.x) | ((unsigned)f2bf(d.y) << 16);
        pd.y = (unsigned)f2bf(d.z) | ((unsigned)f2bf(d.w) << 16);
        *(v2u*)(aw + r * ASTRIDE + c4 * 4)       = po;
        *(v2u*)(aw + r * ASTRIDE + 256 + c4 * 4) = pd;
      }
    }
    asm volatile("s_wait_dscnt 0" ::: "memory");   // A-tile stores visible

    // ---- 16 K-steps x 5 output tiles; manually pipelined LDS fragment fetch ----
    v8f acc[5] = {};
    v4u a0[2], a1[2], b0[2][5], b1[2][5];

    // prologue: issue K-step 0 fragment loads (12 x ds_load_b128)
    DSL(a0[0], aaddr, 0);
    DSL(a1[0], aaddr, 16);
#pragma unroll
    for (int nt = 0; nt < 5; ++nt) {
      DSL(b0[0][nt], baddr[nt], 0);
      DSL(b1[0][nt], baddr[nt], 16);
    }

#pragma unroll
    for (int ks = 0; ks < 16; ++ks) {
      const int cur = ks & 1;
      const int nxt = cur ^ 1;
      // issue next K-step's 12 loads (stay in flight during this step's WMMAs)
      if (ks < 15) {
        DSL(a0[nxt], aaddr, (ks + 1) * 64);
        DSL(a1[nxt], aaddr, (ks + 1) * 64 + 16);
#pragma unroll
        for (int nt = 0; nt < 5; ++nt) {
          DSL(b0[nxt][nt], baddr[nt], (ks + 1) * 64);
          DSL(b1[nxt][nt], baddr[nt], (ks + 1) * 64 + 16);
        }
      }
      // wait until only the 12 just-issued loads are outstanding -> current
      // buffer resident (DS returns in order). "+v" operands make every WMMA
      // depend on this wait and every load complete before it.
#define PIN12(W)                                                         \
      asm volatile(W : "+v"(a0[cur]), "+v"(a1[cur]),                     \
        "+v"(b0[cur][0]), "+v"(b1[cur][0]), "+v"(b0[cur][1]),            \
        "+v"(b1[cur][1]), "+v"(b0[cur][2]), "+v"(b1[cur][2]),            \
        "+v"(b0[cur][3]), "+v"(b1[cur][3]), "+v"(b0[cur][4]),            \
        "+v"(b1[cur][4]))
      if (ks < 15) { PIN12("s_wait_dscnt 0xc"); }
      else         { PIN12("s_wait_dscnt 0x0"); }
#undef PIN12

      union { v16bf v; v4u u[2]; } af;
      af.u[0] = a0[cur];
      af.u[1] = a1[cur];
#pragma unroll
      for (int nt = 0; nt < 5; ++nt) {
        union { v16bf v; v4u u[2]; } bfrag;
        bfrag.u[0] = b0[cur][nt];
        bfrag.u[1] = b1[cur][nt];
        acc[nt] = __builtin_amdgcn_wmma_f32_16x16x32_bf16(
            false, af.v, false, bfrag.v, (short)0, acc[nt], false, false);
      }
    }

    // ---- membrane -> gate: column 0 of tile 4 lives in lanes 0 and 16 ----
    if (lrow == 0) {
#pragma unroll
      for (int r = 0; r < 8; ++r) {
        int row = r + hi * 8;
        float m  = acc[4][r] + bm;
        float mk = mask[(size_t)t * 16 + row];
        float g  = (1.0f / (1.0f + __expf(-m))) * mk;
        gvals[wave][row] = g;
        g_out[(size_t)t * 16 + row] = g;
      }
    }
    asm volatile("s_wait_dscnt 0" ::: "memory");

    // ---- epilogue: e = (acc + b_e) * g, coalesced stores ----
    float gv[8];
#pragma unroll
    for (int r = 0; r < 8; ++r) gv[r] = gvals[wave][r + hi * 8];
#pragma unroll
    for (int nt = 0; nt < 4; ++nt) {
      int col = nt * 16 + lrow;
      float bias = be_s[col];
#pragma unroll
      for (int r = 0; r < 8; ++r) {
        int row = r + hi * 8;
        float val = (acc[nt][r] + bias) * gv[r];
        e_out[((size_t)t * 16 + row) * 64 + col] = val;
      }
    }
  }
}

extern "C" void kernel_launch(void* const* d_in, const int* in_sizes, int n_in,
                              void* d_out, int out_size, void* d_ws, size_t ws_size,
                              hipStream_t stream) {
  const float* obs   = (const float*)d_in[0];
  const float* mask  = (const float*)d_in[1];
  // d_in[2] (incidence, 256 MB) intentionally unused: idx carries the same info
  const int*   idx   = (const int*)d_in[3];
  const float* w_m   = (const float*)d_in[4];
  const float* b_m   = (const float*)d_in[5];
  const float* w_e   = (const float*)d_in[6];
  const float* b_e   = (const float*)d_in[7];

  float* sums   = (float*)d_ws;                       // B*E*D f32 = 8 MB
  float* counts = sums + (size_t)BB * EE * DD;        // B*E f32
  float* g_out  = (float*)d_out;                      // B*N
  float* e_out  = g_out + (size_t)BB * NN;            // B*N*64

  const int nz = BB * EE * DD + BB * EE;
  zero_kernel<<<(nz + 255) / 256, 256, 0, stream>>>(sums, nz);
  scatter_kernel<<<(BB * NN * 64) / 256, 256, 0, stream>>>(obs, idx, sums, counts);
  fused_kernel<<<256, 128, 0, stream>>>(obs, mask, idx, w_m, b_m, w_e, b_e,
                                        sums, counts, g_out, e_out);
}
